// AttentionBlock_12610023981824
// MI455X (gfx1250) — matrix operand order repaired and checked by reading, not compiled, not run on hardware
//
#include <hip/hip_runtime.h>

// ---------------------------------------------------------------------------
// AttentionBlock for MI455X (gfx1250): bf16 WMMA everywhere, flash attention.
// B=4, C=128, L=4096, H=4, hd=32, GROUPS=32.
// Round 2: attention wave owns 2 Q-tiles (32 t rows) -> 2x K/V fragment reuse,
// softmax in exp2 domain (log2e folded into q scale), fast rcp in epilogue.
// ---------------------------------------------------------------------------

#define BATCH 4
#define CCH   128
#define LSEQ  4096
#define NHEAD 4
#define HD    32
#define NGRP  32

typedef __bf16 bf16_t;
typedef bf16_t bf16x8 __attribute__((ext_vector_type(8)));
typedef bf16_t v16bf  __attribute__((ext_vector_type(16)));
typedef float  v8f    __attribute__((ext_vector_type(8)));

__device__ __forceinline__ v16bf combine8(bf16x8 lo, bf16x8 hi) {
  return __builtin_shufflevector(lo, hi, 0,1,2,3,4,5,6,7,8,9,10,11,12,13,14,15);
}

// D(16x16 f32) = A(16x32 bf16) * B(32x16 bf16) + C
__device__ __forceinline__ v8f wmma_bf16(v16bf a, v16bf b, v8f c) {
  return __builtin_amdgcn_wmma_f32_16x16x32_bf16(false, a, false, b, (short)0, c, false, false);
}

// A-fragment: row-major [M, ldk] bf16 source. lane row = lane&15,
// k runs: e0..15 -> k0+16*hf+e
__device__ __forceinline__ v16bf load_a_frag(const bf16_t* base, int ldk,
                                             int row, int k0, int hf) {
  const bf16_t* p = base + (size_t)row * ldk + k0 + hf * 16;
  bf16x8 lo = *(const bf16x8*)(p);
  bf16x8 hi = *(const bf16x8*)(p + 8);
  return combine8(lo, hi);
}

// ---------------------------------------------------------------------------
// Kernel 0: fp32 -> bf16 weight conversion
// ---------------------------------------------------------------------------
__global__ __launch_bounds__(256) void cvt_weights(const float* __restrict__ wq,
                                                   const float* __restrict__ wp,
                                                   bf16_t* __restrict__ wq_b,
                                                   bf16_t* __restrict__ wp_b) {
  int i = blockIdx.x * 256 + threadIdx.x;
  if (i < 3 * CCH * CCH) wq_b[i] = (bf16_t)wq[i];
  if (i < CCH * CCH)     wp_b[i] = (bf16_t)wp[i];
}

// ---------------------------------------------------------------------------
// Kernel 1: GroupNorm, output bf16 hT in (B, L, C) layout
// grid = B*NGRP blocks, 256 threads
// ---------------------------------------------------------------------------
__global__ __launch_bounds__(256) void groupnorm_kernel(const float* __restrict__ x,
                                                        const float* __restrict__ gamma,
                                                        const float* __restrict__ beta,
                                                        bf16_t* __restrict__ hT) {
  const int b = blockIdx.x / NGRP;
  const int g = blockIdx.x % NGRP;
  const int tid = threadIdx.x;
  const int CPG = CCH / NGRP; // 4 channels per group
  const float* xg = x + ((size_t)b * CCH + g * CPG) * LSEQ;

  float s = 0.f, s2 = 0.f;
  for (int c = 0; c < CPG; ++c)
    for (int l = tid; l < LSEQ; l += 256) {
      float v = xg[(size_t)c * LSEQ + l];
      s += v; s2 += v * v;
    }
  __shared__ float red0[256], red1[256];
  red0[tid] = s; red1[tid] = s2;
  __syncthreads();
  for (int off = 128; off; off >>= 1) {
    if (tid < off) { red0[tid] += red0[tid + off]; red1[tid] += red1[tid + off]; }
    __syncthreads();
  }
  const float inv_n = 1.0f / (float)(CPG * LSEQ);
  float mean = red0[0] * inv_n;
  float var  = red1[0] * inv_n - mean * mean;
  float rstd = rsqrtf(var + 1e-5f);

  for (int i = tid; i < CPG * LSEQ; i += 256) {
    int cl = i & (CPG - 1);
    int l  = i >> 2;
    int c  = g * CPG + cl;
    float v = (xg[(size_t)cl * LSEQ + l] - mean) * rstd * gamma[c] + beta[c];
    hT[((size_t)b * LSEQ + l) * CCH + c] = (bf16_t)v;
  }
}

// ---------------------------------------------------------------------------
// Kernel 2: QKV GEMM  qkv[o,l] = sum_c w_qkv[o,c] * hT[l,c]
// Writes: qT (B,H,L,32) scaled by 32^-0.25 * log2(e)  (exp2-domain softmax)
//         kT (B,H,L,32) scaled by 32^-0.25 ; v (B,H,32,L)
// grid = (16, 12, B), 256 threads (8 waves). Wave: 16 M-rows x 64 N-cols.
// ---------------------------------------------------------------------------
__global__ __launch_bounds__(256) void qkv_gemm(const bf16_t* __restrict__ wq,
                                                const bf16_t* __restrict__ hT,
                                                bf16_t* __restrict__ qT,
                                                bf16_t* __restrict__ kT,
                                                bf16_t* __restrict__ vcs) {
  const int w    = threadIdx.x >> 5;
  const int lane = threadIdx.x & 31;
  const int hf   = lane >> 4;
  const int n    = lane & 15;
  const int b    = blockIdx.z;
  const int m0   = blockIdx.y * 32 + (w & 1) * 16;     // 0..368
  const int n0   = blockIdx.x * 256 + (w >> 1) * 64;   // 0..4032
  const bf16_t* hb = hT + (size_t)b * LSEQ * CCH;

  v8f acc[4] = {v8f{}, v8f{}, v8f{}, v8f{}};
  for (int kc = 0; kc < CCH; kc += 32) {
    v16bf a = load_a_frag(wq, CCH, m0 + n, kc, hf);
#pragma unroll
    for (int j = 0; j < 4; ++j) {
      const bf16_t* bp = hb + (size_t)(n0 + j * 16 + n) * CCH + kc + hf * 16;
      v16bf bt = *(const v16bf*)bp;
      acc[j] = wmma_bf16(a, bt, acc[j]);
    }
  }

  const int head = m0 / 96;
  const int rin  = m0 % 96;
  const int sec  = rin / 32;       // 0=q 1=k 2=v
  const int cb   = rin % 32;       // 0 or 16
  const size_t bh = (size_t)(b * NHEAD + head);
  const float kscale = 0.4204482076268573f;                 // 32^-0.25
  const float qscale = 0.4204482076268573f * 1.4426950408889634f; // * log2(e)
  const float mul = (sec == 2) ? 1.0f : (sec == 0 ? qscale : kscale);

#pragma unroll
  for (int j = 0; j < 4; ++j) {
    const int l = n0 + j * 16 + n;
    bf16x8 pk;
#pragma unroll
    for (int r = 0; r < 8; ++r) pk[r] = (bf16_t)(acc[j][r] * mul);
    if (sec == 0) {
      *(bf16x8*)(qT + (bh * LSEQ + l) * HD + cb + 8 * hf) = pk;
    } else if (sec == 1) {
      *(bf16x8*)(kT + (bh * LSEQ + l) * HD + cb + 8 * hf) = pk;
    } else {
#pragma unroll
      for (int r = 0; r < 8; ++r)
        vcs[(bh * HD + cb + 8 * hf + r) * LSEQ + l] = pk[r];
    }
  }
}

// ---------------------------------------------------------------------------
// Kernel 3: flash attention per (b,h). Wave owns TWO 16-row t-tiles (32 rows):
// per 32-wide s-block: 4 WMMA (S) + exp2-domain online softmax + 4 WMMA (PV),
// with K/V fragments reused across both t-tiles.
// grid = (16, H, B), 256 threads (8 waves -> 256 t rows / block).
// ---------------------------------------------------------------------------
__global__ __launch_bounds__(256) void attn_kernel(const bf16_t* __restrict__ qT,
                                                   const bf16_t* __restrict__ kT,
                                                   const bf16_t* __restrict__ vcs,
                                                   bf16_t* __restrict__ oT) {
  __shared__ __align__(32) bf16_t Pbuf[8][2][16][32];  // per-wave, per-tile bounce
  const int w    = threadIdx.x >> 5;
  const int lane = threadIdx.x & 31;
  const int hf   = lane >> 4;
  const int n    = lane & 15;
  const int h    = blockIdx.y;
  const int b    = blockIdx.z;
  const size_t bh = (size_t)(b * NHEAD + h);
  const int t0   = blockIdx.x * 256 + w * 32;

  const bf16_t* q = qT  + bh * LSEQ * HD;
  const bf16_t* k = kT  + bh * LSEQ * HD;
  const bf16_t* v = vcs + bh * HD * LSEQ;

  // Q A-fragments (K = hd = 32 exactly): loaded once per wave
  v16bf qa[2];
  qa[0] = load_a_frag(q, HD, t0 + n, 0, hf);
  qa[1] = load_a_frag(q, HD, t0 + 16 + n, 0, hf);

  v8f Oc[2][2] = {{v8f{}, v8f{}}, {v8f{}, v8f{}}};
  float mrun[2][8], lrun[2][8];
#pragma unroll
  for (int i = 0; i < 2; ++i)
#pragma unroll
    for (int r = 0; r < 8; ++r) { mrun[i][r] = -1e30f; lrun[i][r] = 0.f; }

  for (int s0 = 0; s0 < LSEQ; s0 += 32) {
    // K B-fragments: (32c x 16s), kT is (s, c) with c contiguous
    v16bf kb0 = *(const v16bf*)(k + (size_t)(s0 + n) * HD + hf * 16);
    v16bf kb1 = *(const v16bf*)(k + (size_t)(s0 + 16 + n) * HD + hf * 16);
    v8f S[2][2];
#pragma unroll
    for (int i = 0; i < 2; ++i) {
      S[i][0] = wmma_bf16(qa[i], kb0, v8f{});
      S[i][1] = wmma_bf16(qa[i], kb1, v8f{});
    }

    // online softmax in exp2 domain (log2e already folded into q)
#pragma unroll
    for (int i = 0; i < 2; ++i) {
#pragma unroll
      for (int r = 0; r < 8; ++r) {
        float mx = fmaxf(S[i][0][r], S[i][1][r]);
        mx = fmaxf(mx, __shfl_xor(mx, 1));
        mx = fmaxf(mx, __shfl_xor(mx, 2));
        mx = fmaxf(mx, __shfl_xor(mx, 4));
        mx = fmaxf(mx, __shfl_xor(mx, 8));
        float newm  = fmaxf(mrun[i][r], mx);
        float alpha = __builtin_amdgcn_exp2f(mrun[i][r] - newm);
        mrun[i][r] = newm;
        float p0 = __builtin_amdgcn_exp2f(S[i][0][r] - newm);
        float p1 = __builtin_amdgcn_exp2f(S[i][1][r] - newm);
        float rs = p0 + p1;
        rs += __shfl_xor(rs, 1);
        rs += __shfl_xor(rs, 2);
        rs += __shfl_xor(rs, 4);
        rs += __shfl_xor(rs, 8);
        lrun[i][r] = lrun[i][r] * alpha + rs;
        Oc[i][0][r] *= alpha;
        Oc[i][1][r] *= alpha;
        const int m = r + 8 * hf;            // D-layout row
        Pbuf[w][i][m][n]      = (bf16_t)p0;  // s-local 0..15
        Pbuf[w][i][m][n + 16] = (bf16_t)p1;  // s-local 16..31
      }
    }
    asm volatile("" ::: "memory");           // keep DS store->load order

    // V^T B-fragments (32s x 16c): v is (c, s) with s contiguous
    v16bf vb0 = *(const v16bf*)(v + (size_t)(n)      * LSEQ + s0 + hf * 16);
    v16bf vb1 = *(const v16bf*)(v + (size_t)(16 + n) * LSEQ + s0 + hf * 16);

#pragma unroll
    for (int i = 0; i < 2; ++i) {
      // P A-fragment (16t x 32s) from LDS; DS ops are in-order per wave
      const bf16_t* pb = &Pbuf[w][i][0][0];
      bf16x8 plo = *(const bf16x8*)(pb + n * 32 + hf * 16);
      bf16x8 phi = *(const bf16x8*)(pb + n * 32 + 8 + hf * 16);
      v16bf pa = combine8(plo, phi);
      Oc[i][0] = wmma_bf16(pa, vb0, Oc[i][0]);
      Oc[i][1] = wmma_bf16(pa, vb1, Oc[i][1]);
    }
    asm volatile("" ::: "memory");           // finish loads before next-iter stores
  }

  // epilogue: normalize and write O^T to oT (B, L, C) bf16, c = h*32 + local
  bf16_t* ob = oT + (size_t)b * LSEQ * CCH;
#pragma unroll
  for (int i = 0; i < 2; ++i) {
#pragma unroll
    for (int r = 0; r < 8; ++r) {
      float inv = __builtin_amdgcn_rcpf(lrun[i][r]);
      const int m = r + 8 * hf;
      size_t row = (size_t)(t0 + i * 16 + m) * CCH + h * HD;
      ob[row + n]      = (bf16_t)(Oc[i][0][r] * inv);
      ob[row + 16 + n] = (bf16_t)(Oc[i][1][r] * inv);
    }
  }
}

// ---------------------------------------------------------------------------
// Kernel 4: out[o,l] = sum_c w_proj[o,c] * oT[l,c] + b_proj[o] + x[o,l]
// grid = (16, 4, B), 256 threads (8 waves). Wave: 16 M x 64 N.
// ---------------------------------------------------------------------------
__global__ __launch_bounds__(256) void proj_kernel(const bf16_t* __restrict__ wp,
                                                   const bf16_t* __restrict__ oT,
                                                   const float* __restrict__ bias,
                                                   const float* __restrict__ x,
                                                   float* __restrict__ out) {
  const int w    = threadIdx.x >> 5;
  const int lane = threadIdx.x & 31;
  const int hf   = lane >> 4;
  const int n    = lane & 15;
  const int b    = blockIdx.z;
  const int m0   = blockIdx.y * 32 + (w & 1) * 16;
  const int n0   = blockIdx.x * 256 + (w >> 1) * 64;
  const bf16_t* ob = oT + (size_t)b * LSEQ * CCH;

  v8f acc[4] = {v8f{}, v8f{}, v8f{}, v8f{}};
  for (int kc = 0; kc < CCH; kc += 32) {
    v16bf a = load_a_frag(wp, CCH, m0 + n, kc, hf);
#pragma unroll
    for (int j = 0; j < 4; ++j) {
      const bf16_t* bp = ob + (size_t)(n0 + j * 16 + n) * CCH + kc + hf * 16;
      v16bf bt = *(const v16bf*)bp;
      acc[j] = wmma_bf16(a, bt, acc[j]);
    }
  }
#pragma unroll
  for (int j = 0; j < 4; ++j) {
    const int l = n0 + j * 16 + n;
#pragma unroll
    for (int r = 0; r < 8; ++r) {
      const int oc = m0 + r + 8 * hf;
      const size_t idx = ((size_t)b * CCH + oc) * LSEQ + l;
      out[idx] = acc[j][r] + bias[oc] + x[idx];
    }
  }
}

// ---------------------------------------------------------------------------
// launch
// ---------------------------------------------------------------------------
extern "C" void kernel_launch(void* const* d_in, const int* in_sizes, int n_in,
                              void* d_out, int out_size, void* d_ws, size_t ws_size,
                              hipStream_t stream) {
  const float* x      = (const float*)d_in[0];
  const float* w_qkv  = (const float*)d_in[1];
  const float* w_proj = (const float*)d_in[2];
  const float* b_proj = (const float*)d_in[3];
  const float* gamma  = (const float*)d_in[4];
  const float* beta   = (const float*)d_in[5];
  float* out = (float*)d_out;

  char* ws = (char*)d_ws;
  size_t off = 0;
  bf16_t* wq_b = (bf16_t*)(ws + off); off += (size_t)3 * CCH * CCH * 2;          // 98304
  bf16_t* wp_b = (bf16_t*)(ws + off); off += (size_t)CCH * CCH * 2;              // 32768
  bf16_t* hT   = (bf16_t*)(ws + off); off += (size_t)BATCH * LSEQ * CCH * 2;     // 4 MB
  bf16_t* qT   = (bf16_t*)(ws + off); off += (size_t)BATCH * NHEAD * LSEQ * HD * 2;
  bf16_t* kT   = (bf16_t*)(ws + off); off += (size_t)BATCH * NHEAD * LSEQ * HD * 2;
  bf16_t* vcs  = (bf16_t*)(ws + off); off += (size_t)BATCH * NHEAD * HD * LSEQ * 2;
  bf16_t* oT   = (bf16_t*)(ws + off); off += (size_t)BATCH * LSEQ * CCH * 2;

  cvt_weights<<<(3 * CCH * CCH + 255) / 256, 256, 0, stream>>>(w_qkv, w_proj, wq_b, wp_b);
  groupnorm_kernel<<<BATCH * NGRP, 256, 0, stream>>>(x, gamma, beta, hT);
  qkv_gemm<<<dim3(LSEQ / 256, (3 * CCH) / 32, BATCH), 256, 0, stream>>>(wq_b, hT, qT, kT, vcs);
  attn_kernel<<<dim3(LSEQ / 256, NHEAD, BATCH), 256, 0, stream>>>(qT, kT, vcs, oT);
  proj_kernel<<<dim3(LSEQ / 256, CCH / 32, BATCH), 256, 0, stream>>>(wp_b, oT, b_proj, x, out);
}
